// PaliGemmaWithExpertModel_77421080478474
// MI455X (gfx1250) — matrix operand order repaired and checked by reading, not compiled, not run on hardware
//
#include <hip/hip_runtime.h>
#include <cstdint>
#include <cstddef>

// ---------------------------------------------------------------------------
// CDNA5 (gfx1250) PaliGemma+expert joint transformer forward.
// GEMMs on v_wmma_f32_16x16x32_bf16 (f32 accumulate); activations bf16
// between GEMMs; fp32 weights converted to bf16 while staging through LDS.
// B tiles stored n-major in LDS so WMMA fragment loads are ds_load_b128.
// A tiles staged with GLOBAL_LOAD_ASYNC_TO_LDS_B128 when available.
// ---------------------------------------------------------------------------

typedef __attribute__((ext_vector_type(16))) __bf16 v16bf;
typedef __attribute__((ext_vector_type(8)))  __bf16 v8bf;
typedef __attribute__((ext_vector_type(8)))  float  v8f;
typedef int v4i_n __attribute__((vector_size(16)));  // matches builtin param

#if defined(__AMDGCN__) && __has_builtin(__builtin_amdgcn_global_load_async_to_lds_b128) && __has_builtin(__builtin_amdgcn_s_wait_asynccnt)
#define USE_ASYNC_LDS 1
#else
#define USE_ASYNC_LDS 0
#endif

// ---- problem constants (match reference) ----
constexpr int BB  = 2;
constexpr int PP  = 512;
constexpr int SSX = 64;
constexpr int SS  = PP + SSX;      // 576
constexpr int HH  = 8;
constexpr int HDD = 256;
constexpr int QD  = HH * HDD;      // 2048
constexpr int DD1 = 2048, FF1 = 16384;
constexpr int DD2 = 1024, FF2 = 4096;
constexpr int LLAYERS = 2;
constexpr float ATT_SCALE = 0.0625f;   // HD^-0.5
constexpr float EPSV = 1e-6f;
constexpr float NEGV = -1e9f;

// ---- GEMM tiling ----
constexpr int BM  = 128;
constexpr int BK  = 32;
constexpr int LDA_P = 40;  // LDS A row pitch (bf16): 80B, multiple of 16B
constexpr int BKP   = 40;  // LDS B row pitch (bf16, n-major): 80B

enum { EPI_NONE = 0, EPI_RES = 1, EPI_MASK = 2, EPI_DUALGELU = 3 };

__device__ __forceinline__ float wave_sum(float v) {
  for (int o = 16; o > 0; o >>= 1) v += __shfl_xor(v, o, 32);
  return v;
}
__device__ __forceinline__ float wave_max(float v) {
  for (int o = 16; o > 0; o >>= 1) v = fmaxf(v, __shfl_xor(v, o, 32));
  return v;
}
__device__ __forceinline__ float gelu_tanh(float x) {
  float t = tanhf(0.7978845608028654f * (x + 0.044715f * x * x * x));
  return 0.5f * x * (1.f + t);
}

// ---- stage one BK x BNT B-tile into LDS, n-major: lB[n*BKP + k] ----
template <typename TB, bool BT, int BNT>
__device__ __forceinline__ void stage_B(const TB* __restrict__ Bm,
                                        __bf16* lB, int kb, int bn0, int N,
                                        int K, int ldb, int tid) {
  if constexpr (!BT) {
    if constexpr (sizeof(TB) == 4) {  // fp32 weights -> bf16
      constexpr int N4 = BNT / 4;
      constexpr int ITERS = (BK * N4) / 256;
#pragma unroll
      for (int it = 0; it < ITERS; ++it) {
        int slot = tid + it * 256;
        int k = slot / N4, n4 = slot % N4;
        const float* bp =
            (const float*)Bm + (long long)(kb + k) * ldb + bn0 + n4 * 4;
        float4 f = *reinterpret_cast<const float4*>(bp);
        int nb = n4 * 4;
        lB[(nb + 0) * BKP + k] = (__bf16)f.x;
        lB[(nb + 1) * BKP + k] = (__bf16)f.y;
        lB[(nb + 2) * BKP + k] = (__bf16)f.z;
        lB[(nb + 3) * BKP + k] = (__bf16)f.w;
        if (kb + BK < K) __builtin_prefetch(bp + (long long)BK * ldb, 0, 3);
      }
    } else {  // bf16 B, row-major (attn @ V) -> transpose scatter
      constexpr int N8 = BNT / 8;
      constexpr int ITERS = (BK * N8) / 256;
#pragma unroll
      for (int it = 0; it < ITERS; ++it) {
        int slot = tid + it * 256;
        int k = slot / N8, n8 = slot % N8;
        const __bf16* bp =
            (const __bf16*)Bm + (long long)(kb + k) * ldb + bn0 + n8 * 8;
        v8bf vv = *reinterpret_cast<const v8bf*>(bp);
        int nb = n8 * 8;
#pragma unroll
        for (int e = 0; e < 8; ++e) lB[(nb + e) * BKP + k] = vv[e];
        if (kb + BK < K) __builtin_prefetch(bp + (long long)BK * ldb, 0, 3);
      }
    }
  } else {  // bf16 B accessed transposed (q @ k^T): contiguous along k
    constexpr int ITERS = (BNT * 4) / 256;
#pragma unroll
    for (int it = 0; it < ITERS; ++it) {
      int slot = tid + it * 256;
      int n = slot >> 2, kc = slot & 3;
      int gn = bn0 + n;
      int gnc = (gn < N) ? gn : 0;
      const uint4* bp = reinterpret_cast<const uint4*>(
                            (const __bf16*)Bm + (long long)gnc * ldb + kb) + kc;
      uint4 vv = *bp;
      bool ok = gn < N;
      vv.x = ok ? vv.x : 0u;
      vv.y = ok ? vv.y : 0u;
      vv.z = ok ? vv.z : 0u;
      vv.w = ok ? vv.w : 0u;
      *reinterpret_cast<uint4*>(&lB[n * BKP + kc * 8]) = vv;
      if (kb + BK < K)
        __builtin_prefetch((const __bf16*)bp + BK, 0, 3);
    }
  }
}

// ---------------------------------------------------------------------------
// Tiled WMMA GEMM:  C[M,N] = epilogue( A[M,K](bf16) @ B[K,N] )
//   TB : B type (float weights -> bf16 in LDS, or __bf16 activations)
//   TC : output type (float or __bf16)
//   EPI: none / +residual / *scale+mask / fused gelu-gate (Bm=wg, R=wu)
//   BT : B accessed transposed (B[n*ldb + k])
//   BNT: block N tile (128 normal, 64 for dual-gate)
// 256 threads = 8 waves, wave grid 4(M) x 2(N); wave tile 32 x (BNT/2).
// ---------------------------------------------------------------------------
template <typename TB, typename TC, int EPI, bool BT, int BNT>
__global__ __launch_bounds__(256) void gemm_k(
    const __bf16* __restrict__ A, const TB* __restrict__ Bm,
    TC* __restrict__ C, const float* __restrict__ R,
    int M, int N, int K, int lda, int ldb, int ldc,
    long long sA, long long sB, long long sC, long long sR,
    int zdivB, float alpha, int maskP) {
  constexpr bool DUAL = (EPI == EPI_DUALGELU);
  constexpr int NJ = BNT / 32;  // 16-wide N tiles per wave

  __shared__ __align__(16) __bf16 lA[BM * LDA_P];
  __shared__ __align__(16) __bf16 lB[BNT * BKP];
  __shared__ __align__(16) __bf16 lB2[DUAL ? BNT * BKP : 16];

  const int z = blockIdx.z;
  A += (long long)z * sA;
  Bm += (long long)(z / zdivB) * sB;
  C += (long long)z * sC;
  if constexpr (EPI == EPI_RES) R += (long long)z * sR;
  if constexpr (DUAL) R += (long long)(z / zdivB) * sR;

  const int tid  = threadIdx.x;
  const int lane = tid & 31;
  const int wid  = tid >> 5;
  const int wr   = wid >> 1;  // 0..3 (M)
  const int wc   = wid & 1;   // 0..1 (N)
  const int bm0  = blockIdx.y * BM;
  const int bn0  = blockIdx.x * BNT;

  v8f acc[2][NJ];
  v8f acc2[2][NJ];
#pragma unroll
  for (int i = 0; i < 2; ++i)
#pragma unroll
    for (int j = 0; j < NJ; ++j)
#pragma unroll
      for (int e = 0; e < 8; ++e) {
        acc[i][j][e] = 0.f;
        acc2[i][j][e] = 0.f;
      }

  const int l  = lane & 15;
  const int hi = lane >> 4;

  for (int kb = 0; kb < K; kb += BK) {
    // ---- stage A tile (BM x BK, k-major) ----
#if USE_ASYNC_LDS
    // Async DMA copy global->LDS (16B per lane, ASYNCcnt-tracked). OOB rows
    // are clamped, not zeroed: in a GEMM, C row m depends only on A row m,
    // and the epilogue discards rows gm >= M, so clamped garbage is harmless.
#pragma unroll
    for (int it = 0; it < 2; ++it) {
      int slot = tid + it * 256;         // 512 slots = 128 rows x 4 chunks
      int r = slot >> 2, cs = slot & 3;  // chunk = 8 bf16 = 16B
      int gm = bm0 + r;
      int gmc = (gm < M) ? gm : (M - 1);
      const __bf16* gp = A + (long long)gmc * lda + kb + cs * 8;
      __bf16* lp = &lA[r * LDA_P + cs * 8];
      __builtin_amdgcn_global_load_async_to_lds_b128(
          (__attribute__((address_space(1))) v4i_n*)(unsigned long long)gp,
          (__attribute__((address_space(3))) v4i_n*)lp, 0, 0);
    }
#else
#pragma unroll
    for (int it = 0; it < 2; ++it) {
      int slot = tid + it * 256;         // 512 slots = 128 rows x 4 chunks
      int r = slot >> 2, cs = slot & 3;  // chunk = 8 bf16 = 16B
      int gm = bm0 + r;
      int gmc = (gm < M) ? gm : 0;
      uint4 val =
          *(reinterpret_cast<const uint4*>(A + (long long)gmc * lda + kb) + cs);
      bool ok = gm < M;
      val.x = ok ? val.x : 0u;
      val.y = ok ? val.y : 0u;
      val.z = ok ? val.z : 0u;
      val.w = ok ? val.w : 0u;
      *reinterpret_cast<uint4*>(&lA[r * LDA_P + cs * 8]) = val;
    }
#endif
    // ---- stage B tile(s), n-major ----
    stage_B<TB, BT, BNT>(Bm, lB, kb, bn0, N, K, ldb, tid);
    if constexpr (DUAL)
      stage_B<TB, false, BNT>((const TB*)R, lB2, kb, bn0, N, K, ldb, tid);
#if USE_ASYNC_LDS
    __builtin_amdgcn_s_wait_asynccnt(0);
#endif
    __syncthreads();

    // ---- A fragments (two 16-row tiles) ----
    v16bf af[2];
#pragma unroll
    for (int i = 0; i < 2; ++i) {
      const __bf16* ap = lA + (wr * 32 + i * 16 + l) * LDA_P + 8 * hi;
      v8bf a0 = *reinterpret_cast<const v8bf*>(ap);
      v8bf a1 = *reinterpret_cast<const v8bf*>(ap + 16);
      af[i] = __builtin_shufflevector(a0, a1, 0, 1, 2, 3, 4, 5, 6, 7, 8, 9, 10,
                                      11, 12, 13, 14, 15);
    }
    // ---- B fragments + WMMA ----
#pragma unroll
    for (int j = 0; j < NJ; ++j) {
      const __bf16* bp = lB + (wc * (BNT / 2) + j * 16 + l) * BKP + hi * 8;
      v8bf b0 = *reinterpret_cast<const v8bf*>(bp);
      v8bf b1 = *reinterpret_cast<const v8bf*>(bp + 16);
      v16bf bfc = __builtin_shufflevector(b0, b1, 0, 1, 2, 3, 4, 5, 6, 7, 8, 9,
                                          10, 11, 12, 13, 14, 15);
#pragma unroll
      for (int i = 0; i < 2; ++i)
        acc[i][j] = __builtin_amdgcn_wmma_f32_16x16x32_bf16(
            false, af[i], false, bfc, (short)0, acc[i][j], false, false);
    }
    if constexpr (DUAL) {
#pragma unroll
      for (int j = 0; j < NJ; ++j) {
        const __bf16* bp = lB2 + (wc * (BNT / 2) + j * 16 + l) * BKP + hi * 8;
        v8bf b0 = *reinterpret_cast<const v8bf*>(bp);
        v8bf b1 = *reinterpret_cast<const v8bf*>(bp + 16);
        v16bf bfc = __builtin_shufflevector(b0, b1, 0, 1, 2, 3, 4, 5, 6, 7, 8,
                                            9, 10, 11, 12, 13, 14, 15);
#pragma unroll
        for (int i = 0; i < 2; ++i)
          acc2[i][j] = __builtin_amdgcn_wmma_f32_16x16x32_bf16(
              false, af[i], false, bfc, (short)0, acc2[i][j], false, false);
      }
    }
    __syncthreads();
  }

  // ---- epilogue; C/D layout: VGPR r -> (M = (lane>=16)*8 + r, N = lane&15) ----
  const int n  = lane & 15;
  const int mo = (lane >> 4) * 8;
#pragma unroll
  for (int i = 0; i < 2; ++i) {
#pragma unroll
    for (int j = 0; j < NJ; ++j) {
      int gn = bn0 + wc * (BNT / 2) + j * 16 + n;
#pragma unroll
      for (int r = 0; r < 8; ++r) {
        int gm = bm0 + wr * 32 + i * 16 + mo + r;
        if (gm < M && gn < N) {
          float val = acc[i][j][r];
          float out;
          if constexpr (EPI == EPI_NONE) {
            out = val;
          } else if constexpr (EPI == EPI_RES) {
            out = R[(long long)gm * ldc + gn] + val;
          } else if constexpr (EPI == EPI_MASK) {
            bool ok = (gn < maskP) || (gm >= maskP && gn <= gm);
            out = val * alpha + (ok ? 0.f : NEGV);
          } else {  // EPI_DUALGELU: val = gate, acc2 = up
            out = gelu_tanh(val) * acc2[i][j][r];
          }
          C[(long long)gm * ldc + gn] = (TC)out;
        }
      }
    }
  }
}

// ---------------------------------------------------------------------------
// RMSNorm: y = x * rsqrt(mean(x^2)+eps) * (1 + w)   one block per row
// ---------------------------------------------------------------------------
template <typename TOUT>
__global__ __launch_bounds__(256) void rmsnorm_k(const float* __restrict__ x,
                                                 const float* __restrict__ w,
                                                 TOUT* __restrict__ y, int D) {
  __shared__ float sm[8];
  const int row = blockIdx.x;
  const float* xr = x + (long long)row * D;
  float s = 0.f;
  for (int c = threadIdx.x; c < D; c += 256) {
    float v = xr[c];
    s += v * v;
  }
  s = wave_sum(s);
  if ((threadIdx.x & 31) == 0) sm[threadIdx.x >> 5] = s;
  __syncthreads();
  float tot = 0.f;
#pragma unroll
  for (int i = 0; i < 8; ++i) tot += sm[i];
  float inv = rsqrtf(tot / (float)D + EPSV);
  TOUT* yr = y + (long long)row * D;
  for (int c = threadIdx.x; c < D; c += 256)
    yr[c] = (TOUT)(xr[c] * inv * (1.f + w[c]));
}

// ---------------------------------------------------------------------------
// RoPE + joint-sequence assembly (bf16): q[B,H,S,HD], k[B,S,HD], v[B,S,HD]
// ---------------------------------------------------------------------------
__global__ __launch_bounds__(256) void rope_k(
    const float* __restrict__ qv, const float* __restrict__ qe,
    const float* __restrict__ kv, const float* __restrict__ ke,
    const float* __restrict__ vv, const float* __restrict__ ve,
    __bf16* __restrict__ qo, __bf16* __restrict__ ko, __bf16* __restrict__ vo) {
  const int bs = blockIdx.x;
  const int b = bs / SS, s = bs % SS;
  const int d = threadIdx.x;        // 0..255
  const int j = d & (HDD / 2 - 1);  // d % 128
  const float ang =
      (float)s * expf(-2.f * (float)j / (float)HDD * 9.210340371976184f);
  const float c = cosf(ang), sn = sinf(ang);
  const bool pre = s < PP;
  const long long rowq = pre ? (long long)(b * PP + s) * QD
                             : (long long)(b * SSX + (s - PP)) * QD;
  const long long rowk = pre ? (long long)(b * PP + s) * HDD
                             : (long long)(b * SSX + (s - PP)) * HDD;
  const float* qs = pre ? qv : qe;
  const float* ks = pre ? kv : ke;
  const float* vs = pre ? vv : ve;
  const int d2 = (d < HDD / 2) ? d + HDD / 2 : d - HDD / 2;
  const float sgn = (d < HDD / 2) ? -1.f : 1.f;
#pragma unroll
  for (int h = 0; h < HH; ++h) {
    float x  = qs[rowq + h * HDD + d];
    float xr = sgn * qs[rowq + h * HDD + d2];
    qo[(((long long)b * HH + h) * SS + s) * HDD + d] = (__bf16)(x * c + xr * sn);
  }
  float kx  = ks[rowk + d];
  float kxr = sgn * ks[rowk + d2];
  long long ok = ((long long)b * SS + s) * HDD + d;
  ko[ok] = (__bf16)(kx * c + kxr * sn);
  vo[ok] = (__bf16)vs[rowk + d];
}

// ---------------------------------------------------------------------------
// Row softmax over `cols` columns; writes bf16 probabilities. Block per row.
// ---------------------------------------------------------------------------
__global__ __launch_bounds__(256) void softmax_k(const float* __restrict__ sc,
                                                 __bf16* __restrict__ pr,
                                                 int cols) {
  __shared__ float sm[8];
  __shared__ float ss[8];
  const long long row = blockIdx.x;
  const float* s = sc + row * cols;
  __bf16* p = pr + row * cols;
  float m = -3.4e38f;
  for (int c = threadIdx.x; c < cols; c += 256) m = fmaxf(m, s[c]);
  m = wave_max(m);
  if ((threadIdx.x & 31) == 0) sm[threadIdx.x >> 5] = m;
  __syncthreads();
  float gm = sm[0];
#pragma unroll
  for (int i = 1; i < 8; ++i) gm = fmaxf(gm, sm[i]);
  float su = 0.f;
  for (int c = threadIdx.x; c < cols; c += 256) su += __expf(s[c] - gm);
  su = wave_sum(su);
  if ((threadIdx.x & 31) == 0) ss[threadIdx.x >> 5] = su;
  __syncthreads();
  float tot = 0.f;
#pragma unroll
  for (int i = 0; i < 8; ++i) tot += ss[i];
  float rinv = 1.f / tot;
  for (int c = threadIdx.x; c < cols; c += 256)
    p[c] = (__bf16)(__expf(s[c] - gm) * rinv);
}

// [B,H,S,HD] -> [B,S,H*HD] permute (bf16)
__global__ __launch_bounds__(256) void regroup_k(const __bf16* __restrict__ ai,
                                                 __bf16* __restrict__ ao) {
  long long idx = (long long)blockIdx.x * 256 + threadIdx.x;
  int d = (int)(idx & (HDD - 1));
  long long t = idx >> 8;
  int s = (int)(t % SS);
  long long t2 = t / SS;
  int h = (int)(t2 & (HH - 1));
  int b = (int)(t2 >> 3);
  ao[((long long)b * SS + s) * QD + h * HDD + d] = ai[idx];
}

// ---------------------------------------------------------------------------
extern "C" void kernel_launch(void* const* d_in, const int* in_sizes, int n_in,
                              void* d_out, int out_size, void* d_ws,
                              size_t ws_size, hipStream_t stream) {
  (void)in_sizes; (void)n_in; (void)out_size; (void)ws_size;

  const float* x_vlm   = (const float*)d_in[0];
  const float* x_exp   = (const float*)d_in[1];
  // d_in[2] attention_mask, d_in[3] position_ids: reconstructed analytically.
  const float* vlm_ln1 = (const float*)d_in[4];
  const float* vlm_wq  = (const float*)d_in[5];
  const float* vlm_wk  = (const float*)d_in[6];
  const float* vlm_wv  = (const float*)d_in[7];
  const float* vlm_wo  = (const float*)d_in[8];
  const float* vlm_ln2 = (const float*)d_in[9];
  const float* vlm_wg  = (const float*)d_in[10];
  const float* vlm_wu  = (const float*)d_in[11];
  const float* vlm_wd  = (const float*)d_in[12];
  const float* vlm_nrm = (const float*)d_in[13];
  const float* exp_ln1 = (const float*)d_in[14];
  const float* exp_wq  = (const float*)d_in[15];
  const float* exp_wk  = (const float*)d_in[16];
  const float* exp_wv  = (const float*)d_in[17];
  const float* exp_wo  = (const float*)d_in[18];
  const float* exp_ln2 = (const float*)d_in[19];
  const float* exp_wg  = (const float*)d_in[20];
  const float* exp_wu  = (const float*)d_in[21];
  const float* exp_wd  = (const float*)d_in[22];
  const float* exp_nrm = (const float*)d_in[23];

  const int MV = BB * PP;   // 1024 VLM rows
  const int ME = BB * SSX;  // 128 expert rows

  // ---- workspace bump allocator (256B aligned) ----
  size_t off = 0;
  auto alloc = [&](size_t bytes) -> void* {
    off = (off + 255) & ~(size_t)255;
    void* p = (char*)d_ws + off;
    off += bytes;
    return p;
  };
  float*  xv  = (float*)alloc((size_t)MV * DD1 * 4);
  float*  xe  = (float*)alloc((size_t)ME * DD2 * 4);
  __bf16* hv  = (__bf16*)alloc((size_t)MV * DD1 * 2);
  __bf16* he  = (__bf16*)alloc((size_t)ME * DD2 * 2);
  float*  qvl = (float*)alloc((size_t)MV * QD * 4);
  float*  qel = (float*)alloc((size_t)ME * QD * 4);
  float*  kvl = (float*)alloc((size_t)MV * HDD * 4);
  float*  kel = (float*)alloc((size_t)ME * HDD * 4);
  float*  vvl = (float*)alloc((size_t)MV * HDD * 4);
  float*  vel = (float*)alloc((size_t)ME * HDD * 4);
  __bf16* qbf = (__bf16*)alloc((size_t)BB * HH * SS * HDD * 2);
  __bf16* kbf = (__bf16*)alloc((size_t)BB * SS * HDD * 2);
  __bf16* vbf = (__bf16*)alloc((size_t)BB * SS * HDD * 2);
  float*  scor = (float*)alloc((size_t)BB * HH * SS * SS * 4);
  __bf16* prob = (__bf16*)alloc((size_t)BB * HH * SS * SS * 2);
  __bf16* aob  = (__bf16*)alloc((size_t)BB * HH * SS * HDD * 2);
  __bf16* aor  = (__bf16*)alloc((size_t)BB * SS * QD * 2);
  __bf16* hbv  = (__bf16*)alloc((size_t)MV * FF1 * 2);
  __bf16* hbe  = (__bf16*)alloc((size_t)ME * FF2 * 2);

  (void)hipMemcpyAsync(xv, x_vlm, (size_t)MV * DD1 * 4,
                       hipMemcpyDeviceToDevice, stream);
  (void)hipMemcpyAsync(xe, x_exp, (size_t)ME * DD2 * 4,
                       hipMemcpyDeviceToDevice, stream);

  auto grid = [](int M, int N, int Z, int bnt) {
    return dim3((unsigned)((N + bnt - 1) / bnt), (unsigned)((M + BM - 1) / BM),
                (unsigned)Z);
  };
  const dim3 blk(256);

  for (int l = 0; l < LLAYERS; ++l) {
    const float* w_ln1v = vlm_ln1 + (size_t)l * DD1;
    const float* w_qv   = vlm_wq  + (size_t)l * DD1 * QD;
    const float* w_kv   = vlm_wk  + (size_t)l * DD1 * HDD;
    const float* w_vv   = vlm_wv  + (size_t)l * DD1 * HDD;
    const float* w_ov   = vlm_wo  + (size_t)l * QD * DD1;
    const float* w_ln2v = vlm_ln2 + (size_t)l * DD1;
    const float* w_gv   = vlm_wg  + (size_t)l * DD1 * FF1;
    const float* w_uv   = vlm_wu  + (size_t)l * DD1 * FF1;
    const float* w_dv   = vlm_wd  + (size_t)l * FF1 * DD1;
    const float* w_ln1e = exp_ln1 + (size_t)l * DD2;
    const float* w_qe   = exp_wq  + (size_t)l * DD2 * QD;
    const float* w_ke   = exp_wk  + (size_t)l * DD2 * HDD;
    const float* w_ve   = exp_wv  + (size_t)l * DD2 * HDD;
    const float* w_oe   = exp_wo  + (size_t)l * QD * DD2;
    const float* w_ln2e = exp_ln2 + (size_t)l * DD2;
    const float* w_ge   = exp_wg  + (size_t)l * DD2 * FF2;
    const float* w_ue   = exp_wu  + (size_t)l * DD2 * FF2;
    const float* w_de   = exp_wd  + (size_t)l * FF2 * DD2;

    // ---- pre-attention RMSNorm -> bf16 activations ----
    rmsnorm_k<__bf16><<<MV, blk, 0, stream>>>(xv, w_ln1v, hv, DD1);
    rmsnorm_k<__bf16><<<ME, blk, 0, stream>>>(xe, w_ln1e, he, DD2);

    // ---- QKV projections ----
    gemm_k<float, float, EPI_NONE, false, 128>
        <<<grid(MV, QD, 1, 128), blk, 0, stream>>>(
            hv, w_qv, qvl, nullptr, MV, QD, DD1, DD1, QD, QD, 0, 0, 0, 0, 1,
            1.f, 0);
    gemm_k<float, float, EPI_NONE, false, 128>
        <<<grid(MV, HDD, 1, 128), blk, 0, stream>>>(
            hv, w_kv, kvl, nullptr, MV, HDD, DD1, DD1, HDD, HDD, 0, 0, 0, 0, 1,
            1.f, 0);
    gemm_k<float, float, EPI_NONE, false, 128>
        <<<grid(MV, HDD, 1, 128), blk, 0, stream>>>(
            hv, w_vv, vvl, nullptr, MV, HDD, DD1, DD1, HDD, HDD, 0, 0, 0, 0, 1,
            1.f, 0);
    gemm_k<float, float, EPI_NONE, false, 128>
        <<<grid(ME, QD, 1, 128), blk, 0, stream>>>(
            he, w_qe, qel, nullptr, ME, QD, DD2, DD2, QD, QD, 0, 0, 0, 0, 1,
            1.f, 0);
    gemm_k<float, float, EPI_NONE, false, 128>
        <<<grid(ME, HDD, 1, 128), blk, 0, stream>>>(
            he, w_ke, kel, nullptr, ME, HDD, DD2, DD2, HDD, HDD, 0, 0, 0, 0, 1,
            1.f, 0);
    gemm_k<float, float, EPI_NONE, false, 128>
        <<<grid(ME, HDD, 1, 128), blk, 0, stream>>>(
            he, w_ve, vel, nullptr, ME, HDD, DD2, DD2, HDD, HDD, 0, 0, 0, 0, 1,
            1.f, 0);

    // ---- RoPE + joint sequence assembly ----
    rope_k<<<BB * SS, blk, 0, stream>>>(qvl, qel, kvl, kel, vvl, vel, qbf, kbf,
                                        vbf);

    // ---- scores = q @ k^T * scale + mask (mask analytic) ----
    gemm_k<__bf16, float, EPI_MASK, true, 128>
        <<<grid(SS, SS, BB * HH, 128), blk, 0, stream>>>(
            qbf, kbf, scor, nullptr, SS, SS, HDD, HDD, HDD, SS,
            (long long)SS * HDD, (long long)SS * HDD, (long long)SS * SS, 0, HH,
            ATT_SCALE, PP);

    // ---- softmax -> bf16 probs ----
    softmax_k<<<BB * HH * SS, blk, 0, stream>>>(scor, prob, SS);

    // ---- attn @ V (GQA: single KV head shared by H query heads) ----
    gemm_k<__bf16, __bf16, EPI_NONE, false, 128>
        <<<grid(SS, HDD, BB * HH, 128), blk, 0, stream>>>(
            prob, vbf, aob, nullptr, SS, HDD, SS, SS, HDD, HDD,
            (long long)SS * SS, (long long)SS * HDD, (long long)SS * HDD, 0, HH,
            1.f, 0);

    // ---- [B,H,S,HD] -> [B,S,H*HD] ----
    regroup_k<<<(BB * HH * SS * HDD) / 256, blk, 0, stream>>>(aob, aor);

    // ---- O projection + residual ----
    gemm_k<float, float, EPI_RES, false, 128>
        <<<grid(PP, DD1, BB, 128), blk, 0, stream>>>(
            aor, w_ov, xv, xv, PP, DD1, QD, QD, DD1, DD1, (long long)SS * QD, 0,
            (long long)PP * DD1, (long long)PP * DD1, 1, 1.f, 0);
    gemm_k<float, float, EPI_RES, false, 128>
        <<<grid(SSX, DD2, BB, 128), blk, 0, stream>>>(
            aor + (size_t)PP * QD, w_oe, xe, xe, SSX, DD2, QD, QD, DD2, DD2,
            (long long)SS * QD, 0, (long long)SSX * DD2, (long long)SSX * DD2,
            1, 1.f, 0);

    // ---- MLP: fused gelu(h@wg) * (h@wu) -> bf16, then down-proj + residual ----
    rmsnorm_k<__bf16><<<MV, blk, 0, stream>>>(xv, w_ln2v, hv, DD1);
    gemm_k<float, __bf16, EPI_DUALGELU, false, 64>
        <<<grid(MV, FF1, 1, 64), blk, 0, stream>>>(
            hv, w_gv, hbv, w_uv, MV, FF1, DD1, DD1, FF1, FF1, 0, 0, 0, 0, 1,
            1.f, 0);
    gemm_k<float, float, EPI_RES, false, 128>
        <<<grid(MV, DD1, 1, 128), blk, 0, stream>>>(
            hbv, w_dv, xv, xv, MV, DD1, FF1, FF1, DD1, DD1, 0, 0, 0, 0, 1, 1.f,
            0);

    rmsnorm_k<__bf16><<<ME, blk, 0, stream>>>(xe, w_ln2e, he, DD2);
    gemm_k<float, __bf16, EPI_DUALGELU, false, 64>
        <<<grid(ME, FF2, 1, 64), blk, 0, stream>>>(
            he, w_ge, hbe, w_ue, ME, FF2, DD2, DD2, FF2, FF2, 0, 0, 0, 0, 1,
            1.f, 0);
    gemm_k<float, float, EPI_RES, false, 128>
        <<<grid(ME, DD2, 1, 128), blk, 0, stream>>>(
            hbe, w_de, xe, xe, ME, DD2, FF2, FF2, DD2, DD2, 0, 0, 0, 0, 1, 1.f,
            0);
  }

  // ---- final RMSNorms -> d_out (vlm then exp, fp32) ----
  float* out = (float*)d_out;
  rmsnorm_k<float><<<MV, blk, 0, stream>>>(xv, vlm_nrm, out, DD1);
  rmsnorm_k<float><<<ME, blk, 0, stream>>>(xe, exp_nrm, out + (size_t)MV * DD1,
                                           DD2);
}
